// lstm_seq2seq_69870527972017
// MI455X (gfx1250) — matrix operand order repaired and checked by reading, not compiled, not run on hardware
//
#include <hip/hip_runtime.h>
#include <hip/hip_bf16.h>
#include <stdint.h>

// Problem constants (match reference)
#define B_   1024
#define L_   50
#define D_   4
#define H_   512
#define T_   30
#define G4H  2048          // 4*H gate columns
#define KE   544           // encoder GEMM K: [x(4) | h(512) | pad(28)] -> 17 k-tiles of 32
#define KD   1024          // decoder LSTM K: [comb(512) | h(512)]      -> 32 k-tiles
#define KC   544           // comb GEMM K:    [x(4) | attn(512) | pad]  -> 17 k-tiles

typedef __attribute__((ext_vector_type(16))) __bf16 v16bf;
typedef __attribute__((ext_vector_type(8)))  float  v8f;

union FragU { v16bf v; uint4 q[2]; };

__device__ inline unsigned short f2bf(float f) {              // RNE float->bf16
  union { float f; unsigned int u; } x; x.f = f;
  unsigned int u = x.u;
  u += 0x7fffu + ((u >> 16) & 1u);
  return (unsigned short)(u >> 16);
}
__device__ inline float bf2f(unsigned short s) {
  union { float f; unsigned int u; } x; x.u = ((unsigned int)s) << 16;
  return x.f;
}
__device__ inline float sigm(float x) { return 1.0f / (1.0f + __expf(-x)); }

// A-fragment (16x32 bf16, MxK): lanes 0-15: K = k0..k0+7, k0+16..k0+23 (row M=lane&15)
//                               lanes16-31: K = k0+8..k0+15, k0+24..k0+31
__device__ inline v16bf load_frag_a(const unsigned short* base, int stride, int row0, int k0, int lane) {
  const int half = lane >> 4, r = lane & 15;
  const unsigned short* p = base + (size_t)(row0 + r) * stride + k0 + half * 8;
  FragU f;
  f.q[0] = *(const uint4*)(p);
  f.q[1] = *(const uint4*)(p + 16);
  return f.v;
}
// B-fragment (32x16 bf16, KxN) from row-major W (row = output col n, K contiguous):
// lanes 0-15 hold K = k0..k0+7, k0+16..k0+23 of column n=lane; lanes 16-31 hold K = k0+8..k0+15, k0+24..k0+31.
__device__ inline v16bf load_frag_b(const unsigned short* base, int stride, int n0, int k0, int lane) {
  const int half = lane >> 4, r = lane & 15;
  const unsigned short* p = base + (size_t)(n0 + r) * stride + k0 + half * 8;
  FragU f;
  f.q[0] = *(const uint4*)(p);
  f.q[1] = *(const uint4*)(p + 16);
  return f.v;
}
__device__ inline v8f wmma_bf16(v16bf a, v16bf b, v8f c) {
  return __builtin_amdgcn_wmma_f32_16x16x32_bf16(false, a, false, b, (short)0, c, false, false);
}

// ---------------------------------------------------------------------------
// Pack: build bf16 concatenated/padded weight copies + init xcat pad + x0
// ---------------------------------------------------------------------------
__global__ void pack_kernel(
    const float* __restrict__ eWih, const float* __restrict__ eWhh,
    const float* __restrict__ dWih, const float* __restrict__ dWhh,
    const float* __restrict__ cW,   const float* __restrict__ input,
    unsigned short* __restrict__ WcatE, unsigned short* __restrict__ WcatD,
    unsigned short* __restrict__ combWp, unsigned short* __restrict__ xcat,
    float* __restrict__ xbuf)
{
  const int stride = gridDim.x * blockDim.x;
  const int t0 = blockIdx.x * blockDim.x + threadIdx.x;
  for (int i = t0; i < G4H * KE; i += stride) {          // enc [W_ih | W_hh | 0]
    int n = i / KE, j = i - n * KE;
    float v = 0.f;
    if (j < 4)        v = eWih[n * 4 + j];
    else if (j < 516) v = eWhh[(size_t)n * 512 + (j - 4)];
    WcatE[i] = f2bf(v);
  }
  for (int i = t0; i < G4H * KD; i += stride) {          // dec [W_ih | W_hh]
    int n = i >> 10, j = i & 1023;
    float v = (j < 512) ? dWih[(size_t)n * 512 + j] : dWhh[(size_t)n * 512 + (j - 512)];
    WcatD[i] = f2bf(v);
  }
  for (int i = t0; i < H_ * KC; i += stride) {           // comb_W padded
    int n = i / KC, j = i - n * KC;
    float v = (j < 516) ? cW[(size_t)n * 516 + j] : 0.f;
    combWp[i] = f2bf(v);
  }
  for (int i = t0; i < B_ * KC; i += stride) xcat[i] = 0; // zero (pad cols stay 0)
  for (int i = t0; i < B_ * D_; i += stride) {            // x0 = input[:, L-1, :]
    int b = i >> 2, j = i & 3;
    xbuf[i] = input[((size_t)b * L_ + (L_ - 1)) * D_ + j];
  }
}

// ---------------------------------------------------------------------------
// Encoder: 64 WGs x 16 rows, 16 waves. Wave w owns hidden slice [32w,32w+32):
// accumulators for i/f/g/o tiles and cell state c stay in VGPRs over 50 steps.
// A = [x | h_bf16 | 0] (16 x 544) in LDS, B streamed from L2.
// ---------------------------------------------------------------------------
__global__ __launch_bounds__(512) void enc_kernel(
    const float* __restrict__ x_in, const unsigned short* __restrict__ Wcat,
    const float* __restrict__ b_ih, const float* __restrict__ b_hh,
    unsigned short* __restrict__ encOut, float* __restrict__ h_g,
    float* __restrict__ c_g, unsigned short* __restrict__ dcat)
{
  __shared__ __align__(16) unsigned short A[16 * KE];
  const int r0 = blockIdx.x * 16;
  const int tid = threadIdx.x, wave = tid >> 5, lane = tid & 31;
  const int half = lane >> 4, r = lane & 15;
  const int hh0 = wave * 32;

  for (int i = tid; i < 16 * KE; i += 512) A[i] = 0;     // h0 = 0, pad = 0

  float bias[8];
  #pragma unroll
  for (int g = 0; g < 4; ++g)
    #pragma unroll
    for (int s = 0; s < 2; ++s) {
      int n = g * 512 + hh0 + s * 16 + r;
      bias[g * 2 + s] = b_ih[n] + b_hh[n];
    }

  v8f cc[2];
  #pragma unroll
  for (int s = 0; s < 2; ++s)
    #pragma unroll
    for (int e = 0; e < 8; ++e) cc[s][e] = 0.f;

  __syncthreads();

  #pragma unroll 1
  for (int t = 0; t < L_; ++t) {
    if (tid < 64) {                                      // stage x_t into A[:,0:4]
      int rr = tid >> 2, j = tid & 3;
      A[rr * KE + j] = f2bf(x_in[((size_t)(r0 + rr) * L_ + t) * D_ + j]);
    }
    __syncthreads();

    v8f acc[8];
    #pragma unroll
    for (int q = 0; q < 8; ++q) {
      float bv = bias[q];
      #pragma unroll
      for (int e = 0; e < 8; ++e) acc[q][e] = bv;
    }

    #pragma unroll 1
    for (int kt = 0; kt < KE / 32; ++kt) {
      v16bf av = load_frag_a(A, KE, 0, kt * 32, lane);
      #pragma unroll
      for (int g = 0; g < 4; ++g)
        #pragma unroll
        for (int s = 0; s < 2; ++s) {
          v16bf bv = load_frag_b(Wcat, KE, g * 512 + hh0 + s * 16, kt * 32, lane);
          acc[g * 2 + s] = wmma_bf16(av, bv, acc[g * 2 + s]);
        }
    }
    __syncthreads();                                     // all A reads done

    #pragma unroll
    for (int s = 0; s < 2; ++s) {
      const int hh = hh0 + s * 16 + r;
      #pragma unroll
      for (int v = 0; v < 8; ++v) {
        float iv = acc[0 + s][v], fv = acc[2 + s][v];
        float gv = acc[4 + s][v], ov = acc[6 + s][v];
        float cn = sigm(fv) * cc[s][v] + sigm(iv) * tanhf(gv);
        float hn = sigm(ov) * tanhf(cn);
        cc[s][v] = cn;
        const int m = v + 8 * half;
        const unsigned short hb = f2bf(hn);
        A[m * KE + 4 + hh] = hb;                          // recurrent feed
        encOut[((size_t)(r0 + m) * L_ + t) * H_ + hh] = hb;
        if (t == L_ - 1) {
          size_t gi = (size_t)(r0 + m) * H_ + hh;
          h_g[gi] = hn;
          c_g[gi] = cn;
          dcat[(size_t)(r0 + m) * KD + 512 + hh] = hb;    // decoder h operand
        }
      }
    }
    __syncthreads();
  }
}

// ---------------------------------------------------------------------------
// Attention: per batch element. Reproduces reference's h.reshape(H,-1).T gather:
// h_r[b,k] = h[2k + (b>>9), b&511]. Then softmax over L, weighted sum of
// encoder outputs, writes bf16 [x | attn] row into xcat.
// ---------------------------------------------------------------------------
__global__ __launch_bounds__(64) void attn_kernel(
    const float* __restrict__ xbuf, const float* __restrict__ h_g,
    const float* __restrict__ attn_W, const float* __restrict__ attn_b,
    const unsigned short* __restrict__ encOut, unsigned short* __restrict__ xcat)
{
  const int b = blockIdx.x, tid = threadIdx.x;
  __shared__ float sh[H_];
  __shared__ float sw[L_];
  __shared__ float sx[D_];
  const int bhi = b >> 9, blo = b & 511;
  for (int k = tid; k < H_; k += 64) sh[k] = h_g[(size_t)(2 * k + bhi) * H_ + blo];
  if (tid < D_) sx[tid] = xbuf[b * D_ + tid];
  __syncthreads();

  if (tid < L_) {
    const float* wr = attn_W + (size_t)tid * (H_ + D_);
    float acc = attn_b[tid];
    #pragma unroll
    for (int j = 0; j < D_; ++j) acc += sx[j] * wr[j];
    for (int k = 0; k < H_; ++k) acc += sh[k] * wr[D_ + k];
    sw[tid] = acc;
  }
  __syncthreads();
  float mx = -3.0e38f;
  for (int l = 0; l < L_; ++l) mx = fmaxf(mx, sw[l]);
  float sum = 0.f;
  for (int l = 0; l < L_; ++l) sum += __expf(sw[l] - mx);
  const float inv = 1.f / sum;
  __syncthreads();
  if (tid < L_) sw[tid] = __expf(sw[tid] - mx) * inv;
  __syncthreads();

  for (int hh = tid; hh < H_; hh += 64) {
    float acc = 0.f;
    const unsigned short* e = encOut + (size_t)b * L_ * H_ + hh;
    for (int l = 0; l < L_; ++l) acc += sw[l] * bf2f(e[(size_t)l * H_]);
    xcat[(size_t)b * KC + D_ + hh] = f2bf(acc);
  }
  if (tid < D_) xcat[(size_t)b * KC + tid] = f2bf(sx[tid]);
}

// ---------------------------------------------------------------------------
// comb = relu(xcat @ comb_Wp.T + comb_b) -> bf16 into dcat[:, 0:512]
// Grid (64 M-tiles, 4 col-groups), 8 waves/WG, one 16-col N-tile per wave.
// ---------------------------------------------------------------------------
__global__ __launch_bounds__(256) void comb_kernel(
    const unsigned short* __restrict__ xcat, const unsigned short* __restrict__ Wp,
    const float* __restrict__ comb_b, unsigned short* __restrict__ dcat)
{
  __shared__ __align__(16) unsigned short A[16 * KC];
  const int r0 = blockIdx.x * 16;
  const int tid = threadIdx.x, lane = tid & 31;
  const int half = lane >> 4, r = lane & 15;
  const int n0 = blockIdx.y * 128 + (tid >> 5) * 16;

  {
    const uint4* src = (const uint4*)(xcat + (size_t)r0 * KC);
    uint4* dst = (uint4*)A;
    for (int i = tid; i < 16 * KC / 8; i += 256) dst[i] = src[i];
  }
  const float bv = comb_b[n0 + r];
  __syncthreads();

  v8f acc;
  #pragma unroll
  for (int e = 0; e < 8; ++e) acc[e] = bv;
  #pragma unroll 1
  for (int kt = 0; kt < KC / 32; ++kt) {
    v16bf av = load_frag_a(A, KC, 0, kt * 32, lane);
    v16bf bf = load_frag_b(Wp, KC, n0, kt * 32, lane);
    acc = wmma_bf16(av, bf, acc);
  }
  #pragma unroll
  for (int v = 0; v < 8; ++v) {
    float x = acc[v];
    x = x > 0.f ? x : 0.f;                               // relu
    const int m = v + 8 * half;
    dcat[(size_t)(r0 + m) * KD + n0 + r] = f2bf(x);
  }
}

// ---------------------------------------------------------------------------
// Decoder LSTM: gates = [comb|h] @ [dec_W_ih|dec_W_hh].T + biases (K=1024),
// pointwise update, h -> fp32 master + bf16 back into dcat[:,512:].
// ---------------------------------------------------------------------------
__global__ __launch_bounds__(512) void dec_lstm_kernel(
    const unsigned short* __restrict__ Wcat, const float* __restrict__ b_ih,
    const float* __restrict__ b_hh, unsigned short* __restrict__ dcat,
    float* __restrict__ h_g, float* __restrict__ c_g)
{
  __shared__ __align__(16) unsigned short A[16 * KD];
  const int r0 = blockIdx.x * 16;
  const int tid = threadIdx.x, wave = tid >> 5, lane = tid & 31;
  const int half = lane >> 4, r = lane & 15;
  const int hh0 = wave * 32;

  {
    const uint4* src = (const uint4*)(dcat + (size_t)r0 * KD);
    uint4* dst = (uint4*)A;
    for (int i = tid; i < 16 * KD / 8; i += 512) dst[i] = src[i];
  }

  v8f acc[8];
  #pragma unroll
  for (int g = 0; g < 4; ++g)
    #pragma unroll
    for (int s = 0; s < 2; ++s) {
      const int n = g * 512 + hh0 + s * 16 + r;
      const float bv = b_ih[n] + b_hh[n];
      #pragma unroll
      for (int e = 0; e < 8; ++e) acc[g * 2 + s][e] = bv;
    }
  __syncthreads();

  #pragma unroll 1
  for (int kt = 0; kt < KD / 32; ++kt) {
    v16bf av = load_frag_a(A, KD, 0, kt * 32, lane);
    #pragma unroll
    for (int g = 0; g < 4; ++g)
      #pragma unroll
      for (int s = 0; s < 2; ++s) {
        v16bf bv = load_frag_b(Wcat, KD, g * 512 + hh0 + s * 16, kt * 32, lane);
        acc[g * 2 + s] = wmma_bf16(av, bv, acc[g * 2 + s]);
      }
  }

  #pragma unroll
  for (int s = 0; s < 2; ++s) {
    const int hh = hh0 + s * 16 + r;
    #pragma unroll
    for (int v = 0; v < 8; ++v) {
      const int m = v + 8 * half;
      const size_t gi = (size_t)(r0 + m) * H_ + hh;
      float iv = acc[0 + s][v], fv = acc[2 + s][v];
      float gv = acc[4 + s][v], ov = acc[6 + s][v];
      float cn = sigm(fv) * c_g[gi] + sigm(iv) * tanhf(gv);
      float hn = sigm(ov) * tanhf(cn);
      c_g[gi] = cn;
      h_g[gi] = hn;
      dcat[(size_t)(r0 + m) * KD + 512 + hh] = f2bf(hn);
    }
  }
}

// ---------------------------------------------------------------------------
// Output projection: pred = h @ out_W.T + out_b (1024x512 x 512x4), fp32.
// Writes d_out[b, t, :] and feeds x for the next decoder step.
// ---------------------------------------------------------------------------
__global__ __launch_bounds__(256) void out_kernel(
    const float* __restrict__ h_g, const float* __restrict__ out_W,
    const float* __restrict__ out_b, float* __restrict__ out,
    float* __restrict__ xbuf, int t)
{
  const int idx = blockIdx.x * 256 + threadIdx.x;      // 4096 = B*D
  const int b = idx >> 2, j = idx & 3;
  const float* hr = h_g + (size_t)b * H_;
  const float* w  = out_W + (size_t)j * H_;
  float acc = out_b[j];
  for (int k = 0; k < H_; ++k) acc += hr[k] * w[k];
  out[((size_t)b * T_ + t) * D_ + j] = acc;
  xbuf[b * D_ + j] = acc;
}

// ---------------------------------------------------------------------------
extern "C" void kernel_launch(void* const* d_in, const int* in_sizes, int n_in,
                              void* d_out, int out_size, void* d_ws, size_t ws_size,
                              hipStream_t stream)
{
  (void)in_sizes; (void)n_in; (void)out_size; (void)ws_size;
  const float* input    = (const float*)d_in[0];
  const float* enc_W_ih = (const float*)d_in[2];
  const float* enc_W_hh = (const float*)d_in[3];
  const float* enc_b_ih = (const float*)d_in[4];
  const float* enc_b_hh = (const float*)d_in[5];
  const float* attn_W   = (const float*)d_in[6];
  const float* attn_b   = (const float*)d_in[7];
  const float* comb_W   = (const float*)d_in[8];
  const float* comb_b   = (const float*)d_in[9];
  const float* dec_b_ih = (const float*)d_in[12];
  const float* dec_b_hh = (const float*)d_in[13];
  const float* out_W    = (const float*)d_in[14];
  const float* out_b    = (const float*)d_in[15];
  const float* dec_W_ih = (const float*)d_in[10];
  const float* dec_W_hh = (const float*)d_in[11];

  uint8_t* ws = (uint8_t*)d_ws;
  size_t off = 0;
  auto take = [&](size_t bytes) -> void* {
    void* p = ws + off;
    off += (bytes + 255) & ~(size_t)255;
    return p;
  };
  unsigned short* WcatE  = (unsigned short*)take((size_t)G4H * KE * 2);   // 2.2 MB
  unsigned short* WcatD  = (unsigned short*)take((size_t)G4H * KD * 2);   // 4.2 MB
  unsigned short* combWp = (unsigned short*)take((size_t)H_  * KC * 2);   // 0.6 MB
  unsigned short* encOut = (unsigned short*)take((size_t)B_ * L_ * H_ * 2); // 52 MB
  float*          h_g    = (float*)take((size_t)B_ * H_ * 4);             // 2 MB
  float*          c_g    = (float*)take((size_t)B_ * H_ * 4);             // 2 MB
  unsigned short* dcat   = (unsigned short*)take((size_t)B_ * KD * 2);    // 2 MB
  unsigned short* xcat   = (unsigned short*)take((size_t)B_ * KC * 2);    // 1.1 MB
  float*          xbuf   = (float*)take((size_t)B_ * D_ * 4);

  pack_kernel<<<1024, 256, 0, stream>>>(enc_W_ih, enc_W_hh, dec_W_ih, dec_W_hh,
                                        comb_W, input, WcatE, WcatD, combWp, xcat, xbuf);
  enc_kernel<<<B_ / 16, 512, 0, stream>>>(input, WcatE, enc_b_ih, enc_b_hh,
                                          encOut, h_g, c_g, dcat);
  float* out_f = (float*)d_out;
  for (int t = 0; t < T_; ++t) {
    attn_kernel<<<B_, 64, 0, stream>>>(xbuf, h_g, attn_W, attn_b, encOut, xcat);
    comb_kernel<<<dim3(B_ / 16, 4), 256, 0, stream>>>(xcat, combWp, comb_b, dcat);
    dec_lstm_kernel<<<B_ / 16, 512, 0, stream>>>(WcatD, dec_b_ih, dec_b_hh, dcat, h_g, c_g);
    out_kernel<<<B_ * D_ / 256, 256, 0, stream>>>(h_g, out_W, out_b, out_f, xbuf, t);
  }
}
